// GatedDeltaNet_88330297410073
// MI455X (gfx1250) — matrix operand order repaired and checked by reading, not compiled, not run on hardware
//
#include <hip/hip_runtime.h>
#include <cstdint>
#include <cstddef>

// ---------------------------------------------------------------------------
// Types
// ---------------------------------------------------------------------------
typedef __attribute__((ext_vector_type(16))) __bf16 v16bf;
typedef __attribute__((ext_vector_type(8)))  float  v8f;
typedef __attribute__((ext_vector_type(8)))  __bf16 bf16x8;

// ---------------------------------------------------------------------------
// Optional gfx1250 async global->LDS path (guarded; safe fallback otherwise)
// ROCm 7.2 signature: (v4i* gsrc, v4i* ldsdst, imm offset, imm cpol) with
// generic pointers to 16-byte int vectors (probe-confirmed by diagnostics).
// ---------------------------------------------------------------------------
#if defined(__has_builtin)
#  if __has_builtin(__builtin_amdgcn_global_load_async_to_lds_b128)
#    define ASYNC_LDS 1
#  endif
#  if __has_builtin(__builtin_amdgcn_s_wait_asynccnt)
#    define HAVE_WAIT_ASYNC_BUILTIN 1
#  endif
#endif

#if defined(ASYNC_LDS)
typedef int v4i_vs __attribute__((vector_size(4 * sizeof(int))));
static __device__ inline void async_copy16(const void* g, void* l) {
  __builtin_amdgcn_global_load_async_to_lds_b128((v4i_vs*)g, (v4i_vs*)l, 0, 0);
}
#endif

static __device__ inline void wait_async0() {
#if defined(HAVE_WAIT_ASYNC_BUILTIN)
  __builtin_amdgcn_s_wait_asynccnt(0);
#elif defined(ASYNC_LDS)
  asm volatile("s_wait_asynccnt 0" ::: "memory");
#endif
}

// ---------------------------------------------------------------------------
// Problem dimensions (fixed by the reference)
// ---------------------------------------------------------------------------
constexpr int Bc   = 2;
constexpr int Sseq = 4096;
constexpr int HIDc = 2048;
constexpr int HVc  = 32;
constexpr int DKc  = 128;
constexpr int DVc  = 128;
constexpr int CH   = 64;                  // chunk length
constexpr int KEYD = 16 * DKc;            // 2048
constexpr int VALD = HVc * DVc;           // 4096
constexpr int CDIM = 2 * KEYD + VALD;     // 8192
constexpr int MTOK = Bc * Sseq;           // 8192
constexpr float EPSc = 1e-6f;
constexpr float RSQRT_DK = 0.088388347648318447f;  // 1/sqrt(128)

#define WMMA_BF16(c, a, b) \
  (c) = __builtin_amdgcn_wmma_f32_16x16x32_bf16(false, (a), false, (b), (short)0, (c), false, false)

// ---------------------------------------------------------------------------
// WMMA fragment loaders (contiguous k order, shared by A and B).
// A 16x32 bf16 frag: lane = m + 16*(k>>4), element e <-> k = e + 16*hi.
// B 32x16 bf16 frag: lane = n + 16*(k>>4), element e <-> k = e + 16*khi.
// C 16x16 f32 frag: element i <-> m = i + 8*(lane>>4), n = lane&15.
// f32 variants read f32 LDS and convert to bf16 inline (transpose and row/k
// scaling folded into the load for free).
// ---------------------------------------------------------------------------
static __device__ inline v16bf frag_a_f32(const float* A, int lda, int m0, int k0,
                                          const float* rscale) {
  const int lane = threadIdx.x & 31;
  const int ml = m0 + (lane & 15);
  const int hi = (lane >> 4) & 1;
  const float rs = rscale ? rscale[ml] : 1.0f;
  v16bf a;
#pragma unroll
  for (int e = 0; e < 16; ++e) {
    int k = e + 16 * hi;
    a[e] = (__bf16)(A[(size_t)ml * lda + k0 + k] * rs);
  }
  return a;
}

// logical A(m,k) = src[k][m] * kscale[k]   (transposed source, k-indexed scale)
static __device__ inline v16bf frag_aT_f32(const float* A, int lda, int m0, int k0,
                                           const float* kscale) {
  const int lane = threadIdx.x & 31;
  const int ml = m0 + (lane & 15);
  const int hi = (lane >> 4) & 1;
  v16bf a;
#pragma unroll
  for (int e = 0; e < 16; ++e) {
    int k = k0 + e + 16 * hi;
    float s = kscale ? kscale[k] : 1.0f;
    a[e] = (__bf16)(A[(size_t)k * lda + ml] * s);
  }
  return a;
}

static __device__ inline v16bf frag_b_f32(const float* Bm, int ldb, int k0, int n0,
                                          const float* kscale) {
  const int lane = threadIdx.x & 31;
  const int nl = n0 + (lane & 15);
  const int khi = (lane >> 4) & 1;
  v16bf b;
#pragma unroll
  for (int e = 0; e < 16; ++e) {
    int k = k0 + e + 16 * khi;
    float s = kscale ? kscale[k] : 1.0f;
    b[e] = (__bf16)(Bm[(size_t)k * ldb + nl] * s);
  }
  return b;
}

// logical B(k,n) = src[n][k]   (use a row-major matrix as its transpose)
static __device__ inline v16bf frag_bT_f32(const float* Bm, int ldb, int k0, int n0) {
  const int lane = threadIdx.x & 31;
  const int nl = n0 + (lane & 15);
  const int khi = (lane >> 4) & 1;
  v16bf b;
#pragma unroll
  for (int e = 0; e < 16; ++e) {
    int k = k0 + e + 16 * khi;
    b[e] = (__bf16)(Bm[(size_t)nl * ldb + k]);
  }
  return b;
}

// bf16-LDS variants for the big GEMM (no conversion; fully contiguous loads)
static __device__ inline v16bf frag_a_bf(const __bf16* A, int lda, int m0, int k0) {
  const int lane = threadIdx.x & 31;
  const int ml = m0 + (lane & 15);
  const int hi = (lane >> 4) & 1;
  v16bf a;
#pragma unroll
  for (int e = 0; e < 16; ++e) {
    int k = e + 16 * hi;
    a[e] = A[(size_t)ml * lda + k0 + k];
  }
  return a;
}
// B tile stored TRANSPOSED in LDS: BmT[n][k] -> every lane reads 16 contiguous
// bf16 (two ds_load_b128) instead of 16 strided u16 loads.
static __device__ inline v16bf frag_b_bfT(const __bf16* BmT, int ldbt, int k0, int n0) {
  const int lane = threadIdx.x & 31;
  const int nl = n0 + (lane & 15);
  const int khi = (lane >> 4) & 1;
  v16bf b;
#pragma unroll
  for (int e = 0; e < 16; ++e)
    b[e] = BmT[(size_t)nl * ldbt + k0 + 16 * khi + e];
  return b;
}

// ---------------------------------------------------------------------------
// Kernel: f32 -> bf16 conversion (4 elements / thread)
// ---------------------------------------------------------------------------
__global__ __launch_bounds__(256) void cvt_f32_bf16_kernel(const float* __restrict__ in,
                                                           __bf16* __restrict__ out,
                                                           size_t n) {
  size_t base = ((size_t)blockIdx.x * blockDim.x + threadIdx.x) * 4;
  if (base + 3 < n) {
#pragma unroll
    for (int j = 0; j < 4; ++j) out[base + j] = (__bf16)in[base + j];
  } else {
    for (size_t i = base; i < n; ++i) out[i] = (__bf16)in[i];
  }
}

// ---------------------------------------------------------------------------
// Kernel: bf16 WMMA GEMM.  C[M,N] = A[M,K] * W[N,K]^T   (W row-major (N,K))
// Block: 256 threads (8 waves). Tile: 128x128, K-step 32.
// A and B tiles both staged as per-lane 16B contiguous rows; staging uses
// GLOBAL_LOAD_ASYNC_TO_LDS_B128 with LDS double buffering when available
// (wait -> barrier -> issue tile t+1 -> WMMA on tile t).
// ---------------------------------------------------------------------------
constexpr int GBM = 128, GBN = 128, GBK = 32;

__global__ __launch_bounds__(256) void gemm_bf16_kernel(const __bf16* __restrict__ A,
                                                        const __bf16* __restrict__ W,
                                                        float* __restrict__ Cf,
                                                        __bf16* __restrict__ Cb,
                                                        int M, int N, int K,
                                                        int outIsBf16) {
  __shared__ __align__(16) __bf16 As[2][GBM][GBK + 8];
  __shared__ __align__(16) __bf16 Bs[2][GBN][GBK + 8];   // transposed: [n][k]

  const int tid = threadIdx.x;
  const int wid = tid >> 5;
  const int lane = tid & 31;
  const int hihalf = lane >> 4, nloc = lane & 15;
  const int m_blk = blockIdx.y * GBM;
  const int n_blk = blockIdx.x * GBN;
  const int wr = (wid >> 2) * 64;        // 0 or 64
  const int wc = (wid & 3) * 32;         // 0,32,64,96

  v8f acc[4][2] = {};

  // A and B staging have identical shape: row r (128 rows), 8-bf16 chunk cc.
  auto stage_tile = [&](int buf, int k0) {
#pragma unroll
    for (int it = 0; it < 2; ++it) {
      int chunk = tid + it * 256;        // 0..511
      int r = chunk >> 2;
      int cc = (chunk & 3) * 8;
#if defined(ASYNC_LDS)
      async_copy16(&A[(size_t)(m_blk + r) * K + k0 + cc], &As[buf][r][cc]);
      async_copy16(&W[(size_t)(n_blk + r) * K + k0 + cc], &Bs[buf][r][cc]);
#else
      *(bf16x8*)&As[buf][r][cc] = *(const bf16x8*)&A[(size_t)(m_blk + r) * K + k0 + cc];
      *(bf16x8*)&Bs[buf][r][cc] = *(const bf16x8*)&W[(size_t)(n_blk + r) * K + k0 + cc];
#endif
    }
  };

  auto compute_tile = [&](int buf) {
    v16bf bfr0 = frag_b_bfT(&Bs[buf][0][0], GBK + 8, 0, wc);
    v16bf bfr1 = frag_b_bfT(&Bs[buf][0][0], GBK + 8, 0, wc + 16);
#pragma unroll
    for (int mt = 0; mt < 4; ++mt) {
      v16bf afr = frag_a_bf(&As[buf][0][0], GBK + 8, wr + mt * 16, 0);
      WMMA_BF16(acc[mt][0], afr, bfr0);
      WMMA_BF16(acc[mt][1], afr, bfr1);
    }
  };

#if defined(ASYNC_LDS)
  stage_tile(0, 0);
  int cur = 0;
  for (int k0 = 0; k0 < K; k0 += GBK) {
    wait_async0();
    __syncthreads();                 // tile `cur` resident; prior reads done
    if (k0 + GBK < K) stage_tile(cur ^ 1, k0 + GBK);  // overlap with WMMA
    compute_tile(cur);
    cur ^= 1;
  }
#else
  for (int k0 = 0; k0 < K; k0 += GBK) {
    stage_tile(0, k0);
    if (k0 + GBK < K) {              // -> global_prefetch_b8
      __builtin_prefetch(&A[(size_t)(m_blk + (tid >> 1)) * K + k0 + GBK]);
      __builtin_prefetch(&W[(size_t)(n_blk + (tid >> 1)) * K + k0 + GBK]);
    }
    __syncthreads();
    compute_tile(0);
    __syncthreads();
  }
#endif

#pragma unroll
  for (int mt = 0; mt < 4; ++mt)
#pragma unroll
    for (int nt = 0; nt < 2; ++nt)
#pragma unroll
      for (int i = 0; i < 8; ++i) {
        int m = m_blk + wr + mt * 16 + i + 8 * hihalf;
        int n = n_blk + wc + nt * 16 + nloc;
        float v = acc[mt][nt][i];
        if (outIsBf16) Cb[(size_t)m * N + n] = (__bf16)v;
        else           Cf[(size_t)m * N + n] = v;
      }
}

// ---------------------------------------------------------------------------
// Kernel: fused b/a projections (N=32 each) + beta = sigmoid(b),
//         g = -exp(A_log) * softplus(a + dt_bias).   One block per token.
// ---------------------------------------------------------------------------
__global__ __launch_bounds__(256) void proj_ba_kernel(const float* __restrict__ hs,
                                                      const float* __restrict__ wb,
                                                      const float* __restrict__ wa,
                                                      const float* __restrict__ dt_bias,
                                                      const float* __restrict__ A_log,
                                                      float* __restrict__ beta_out,
                                                      float* __restrict__ g_out) {
  const int tok = blockIdx.x;
  const float* hrow = hs + (size_t)tok * HIDc;
  const int n = threadIdx.x & 31;
  const int grp = threadIdx.x >> 5;
  float accB = 0.f, accA = 0.f;
  const int kb = grp * (HIDc / 8);
  for (int k = kb; k < kb + HIDc / 8; ++k) {
    float hv = hrow[k];
    accB += hv * wb[(size_t)n * HIDc + k];
    accA += hv * wa[(size_t)n * HIDc + k];
  }
  __shared__ float redB[8][32];
  __shared__ float redA[8][32];
  redB[grp][n] = accB;
  redA[grp][n] = accA;
  __syncthreads();
  if (threadIdx.x < 32) {
    float sb = 0.f, sa = 0.f;
#pragma unroll
    for (int g = 0; g < 8; ++g) { sb += redB[g][n]; sa += redA[g][n]; }
    float beta = 1.f / (1.f + __expf(-sb));
    float aa = sa + dt_bias[n];
    float sp = (aa > 20.f) ? aa : log1pf(__expf(aa));
    float gg = -__expf(A_log[n]) * sp;
    beta_out[(size_t)tok * HVc + n] = beta;
    g_out[(size_t)tok * HVc + n] = gg;
  }
}

// ---------------------------------------------------------------------------
// Kernel: causal depthwise conv (K=4) + SiLU.  8 timesteps per thread.
// ---------------------------------------------------------------------------
__global__ __launch_bounds__(256) void conv_silu_kernel(const __bf16* __restrict__ qkv,
                                                        const float* __restrict__ convw,
                                                        __bf16* __restrict__ xout) {
  size_t gid = (size_t)blockIdx.x * blockDim.x + threadIdx.x;
  int c = (int)(gid % CDIM);
  size_t t2 = gid / CDIM;
  int sblk = (int)(t2 % (Sseq / 8));
  int b = (int)(t2 / (Sseq / 8));
  if (b >= Bc) return;
  float w0 = convw[c * 4 + 0], w1 = convw[c * 4 + 1];
  float w2 = convw[c * 4 + 2], w3 = convw[c * 4 + 3];
  int s0 = sblk * 8;
  float buf[11];
#pragma unroll
  for (int j = 0; j < 11; ++j) {
    int s = s0 - 3 + j;
    buf[j] = (s >= 0) ? (float)qkv[((size_t)(b * Sseq + s)) * CDIM + c] : 0.f;
  }
#pragma unroll
  for (int i = 0; i < 8; ++i) {
    float a = buf[i] * w0 + buf[i + 1] * w1 + buf[i + 2] * w2 + buf[i + 3] * w3;
    float sig = 1.f / (1.f + __expf(-a));
    xout[((size_t)(b * Sseq + s0 + i)) * CDIM + c] = (__bf16)(a * sig);
  }
}

// ---------------------------------------------------------------------------
// Kernel: chunked gated delta rule core.  One workgroup per (b, head).
// 8 waves, all matmuls via v_wmma_f32_16x16x32_bf16 on LDS-resident tiles.
// Triangular inverse (I-A)^{-1} via nilpotent doubling (5 WMMA squarings).
// ---------------------------------------------------------------------------
__global__ __launch_bounds__(256) void core_kernel(const __bf16* __restrict__ xbf,
                                                   const float* __restrict__ g_all,
                                                   const float* __restrict__ beta_all,
                                                   float* __restrict__ core_out) {
  const int bh = blockIdx.x;
  const int b = bh / HVc, h = bh % HVc;
  const int hk = h >> 1;  // HV/HK = 2
  const int tid = threadIdx.x, wid = tid >> 5, lane = tid & 31;
  const int hihalf = lane >> 4, nloc = lane & 15;

  __shared__ float Sst[DKc][DVc + 1];    // 128x129 f32 state
  __shared__ float kmat[CH][DKc + 1];
  __shared__ float qmat[CH][DKc + 1];
  __shared__ float vmat[CH][DKc + 1];    // v -> u -> v_new (in place)
  __shared__ float wmat[CH][DKc + 1];
  __shared__ float Pm[CH][CH + 1];       // (I-A)^{-1}
  __shared__ float Mm[CH][CH + 1];       // A^(2^t)
  __shared__ float Tm[CH][CH + 1];       // scratch / attn_c
  __shared__ float Dm[CH][CH + 1];       // decay mask (tril incl diag)
  __shared__ float gcum[CH];
  __shared__ float betar[CH];
  __shared__ float scl1[CH];
  __shared__ float scl2[CH];

  for (int i = tid; i < DKc * (DVc + 1); i += 256) (&Sst[0][0])[i] = 0.f;
  __syncthreads();

  for (int nc = 0; nc < Sseq / CH; ++nc) {
    const int sbase = nc * CH;

    // ---- gate cumsum + beta -------------------------------------------------
    if (tid == 0) {
      float acc = 0.f;
      for (int i = 0; i < CH; ++i) {
        acc += g_all[((size_t)(b * Sseq + sbase + i)) * HVc + h];
        gcum[i] = acc;
      }
    }
    if (tid < CH)
      betar[tid] = beta_all[((size_t)(b * Sseq + sbase + tid)) * HVc + h];
    __syncthreads();

    // ---- load q,k,v chunk ---------------------------------------------------
    for (int idx = tid; idx < CH * DKc; idx += 256) {
      int c = idx >> 7, d = idx & 127;
      const __bf16* row = xbf + ((size_t)(b * Sseq + sbase + c)) * CDIM;
      qmat[c][d] = (float)row[hk * DKc + d];
      kmat[c][d] = (float)row[KEYD + hk * DKc + d];
      vmat[c][d] = (float)row[2 * KEYD + h * DVc + d];
    }
    __syncthreads();
    // l2norm scales
    if (tid < CH) {
      float sq = 0.f, sk = 0.f;
      for (int d = 0; d < DKc; ++d) {
        float a = qmat[tid][d]; sq += a * a;
        float e = kmat[tid][d]; sk += e * e;
      }
      scl1[tid] = rsqrtf(sq + EPSc) * RSQRT_DK;
      scl2[tid] = rsqrtf(sk + EPSc);
    }
    __syncthreads();
    for (int idx = tid; idx < CH * DKc; idx += 256) {
      int c = idx >> 7, d = idx & 127;
      qmat[c][d] *= scl1[c];
      kmat[c][d] *= scl2[c];
    }
    // decay mask Dm[i][j] = (j<=i) ? exp(gcum[i]-gcum[j]) : 0
    for (int idx = tid; idx < CH * CH; idx += 256) {
      int i = idx >> 6, j = idx & 63;
      Dm[i][j] = (j <= i) ? __expf(gcum[i] - gcum[j]) : 0.f;
    }
    __syncthreads();

    // ---- attn = -(kb @ k^T) * decay (strict); Pm = I + attn; Mm = attn -----
    {
#pragma unroll
      for (int t = 0; t < 2; ++t) {
        int tt = wid * 2 + t;
        int tm = (tt >> 2) * 16, tn = (tt & 3) * 16;
        v8f acc = {};
#pragma unroll
        for (int k0 = 0; k0 < DKc; k0 += 32) {
          v16bf a = frag_a_f32(&kmat[0][0], DKc + 1, tm, k0, betar);
          v16bf bb = frag_bT_f32(&kmat[0][0], DKc + 1, k0, tn);
          WMMA_BF16(acc, a, bb);
        }
#pragma unroll
        for (int i = 0; i < 8; ++i) {
          int m = tm + i + 8 * hihalf, n = tn + nloc;
          float sv = (n < m) ? -acc[i] * Dm[m][n] : 0.f;
          Mm[m][n] = sv;
          Pm[m][n] = sv + ((m == n) ? 1.f : 0.f);
        }
      }
    }
    __syncthreads();

    // ---- (I - A)^{-1} via doubling: 5x { M = M@M; P += P@M } ---------------
    for (int it = 0; it < 5; ++it) {
      // Tm = Mm @ Mm
#pragma unroll
      for (int t = 0; t < 2; ++t) {
        int tt = wid * 2 + t;
        int tm = (tt >> 2) * 16, tn = (tt & 3) * 16;
        v8f acc = {};
#pragma unroll
        for (int k0 = 0; k0 < CH; k0 += 32) {
          v16bf a = frag_a_f32(&Mm[0][0], CH + 1, tm, k0, nullptr);
          v16bf bb = frag_b_f32(&Mm[0][0], CH + 1, k0, tn, nullptr);
          WMMA_BF16(acc, a, bb);
        }
#pragma unroll
        for (int i = 0; i < 8; ++i)
          Tm[tm + i + 8 * hihalf][tn + nloc] = acc[i];
      }
      __syncthreads();
      // delta = Pm @ Tm  (read phase), then Pm += delta, Mm = Tm
      v8f accP[2];
#pragma unroll
      for (int t = 0; t < 2; ++t) {
        int tt = wid * 2 + t;
        int tm = (tt >> 2) * 16, tn = (tt & 3) * 16;
        v8f acc = {};
#pragma unroll
        for (int k0 = 0; k0 < CH; k0 += 32) {
          v16bf a = frag_a_f32(&Pm[0][0], CH + 1, tm, k0, nullptr);
          v16bf bb = frag_b_f32(&Tm[0][0], CH + 1, k0, tn, nullptr);
          WMMA_BF16(acc, a, bb);
        }
        accP[t] = acc;
      }
      __syncthreads();
#pragma unroll
      for (int t = 0; t < 2; ++t) {
        int tt = wid * 2 + t;
        int tm = (tt >> 2) * 16, tn = (tt & 3) * 16;
#pragma unroll
        for (int i = 0; i < 8; ++i)
          Pm[tm + i + 8 * hihalf][tn + nloc] += accP[t][i];
      }
      for (int idx = tid; idx < CH * CH; idx += 256) {
        int i = idx >> 6, j = idx & 63;
        Mm[i][j] = Tm[i][j];
      }
      __syncthreads();
    }

    // scales for u/w and the output phase
    if (tid < CH) {
      scl1[tid] = betar[tid] * __expf(gcum[tid]);  // kb * exp(g) row scale
      scl2[tid] = __expf(gcum[tid]);               // exp(g) row scale for q
    }
    __syncthreads();

    // ---- u = T @ (v*beta) -> vmat ; w = T @ (kb*exp(g)) -> wmat ------------
    {
      v8f accU[4] = {}, accW[4] = {};
#pragma unroll
      for (int t = 0; t < 4; ++t) {
        int tt = wid * 4 + t;
        int tm = (tt >> 3) * 16, tn = (tt & 7) * 16;
#pragma unroll
        for (int k0 = 0; k0 < CH; k0 += 32) {
          v16bf a = frag_a_f32(&Pm[0][0], CH + 1, tm, k0, nullptr);
          v16bf bu = frag_b_f32(&vmat[0][0], DKc + 1, k0, tn, betar);
          v16bf bw = frag_b_f32(&kmat[0][0], DKc + 1, k0, tn, scl1);
          WMMA_BF16(accU[t], a, bu);
          WMMA_BF16(accW[t], a, bw);
        }
      }
      __syncthreads();
#pragma unroll
      for (int t = 0; t < 4; ++t) {
        int tt = wid * 4 + t;
        int tm = (tt >> 3) * 16, tn = (tt & 7) * 16;
#pragma unroll
        for (int i = 0; i < 8; ++i) {
          int m = tm + i + 8 * hihalf, n = tn + nloc;
          vmat[m][n] = accU[t][i];
          wmat[m][n] = accW[t][i];
        }
      }
    }
    __syncthreads();

    // ---- v_new = u - w @ S  (in place in vmat) -----------------------------
    {
      v8f accV[4] = {};
#pragma unroll
      for (int t = 0; t < 4; ++t) {
        int tt = wid * 4 + t;
        int tm = (tt >> 3) * 16, tn = (tt & 7) * 16;
#pragma unroll
        for (int k0 = 0; k0 < DKc; k0 += 32) {
          v16bf a = frag_a_f32(&wmat[0][0], DKc + 1, tm, k0, nullptr);
          v16bf bb = frag_b_f32(&Sst[0][0], DVc + 1, k0, tn, nullptr);
          WMMA_BF16(accV[t], a, bb);
        }
      }
      __syncthreads();
#pragma unroll
      for (int t = 0; t < 4; ++t) {
        int tt = wid * 4 + t;
        int tm = (tt >> 3) * 16, tn = (tt & 7) * 16;
#pragma unroll
        for (int i = 0; i < 8; ++i) {
          int m = tm + i + 8 * hihalf, n = tn + nloc;
          vmat[m][n] = vmat[m][n] - accV[t][i];
        }
      }
    }
    __syncthreads();

    // ---- attn_c = (q @ k^T) * decay -> Tm ----------------------------------
    {
#pragma unroll
      for (int t = 0; t < 2; ++t) {
        int tt = wid * 2 + t;
        int tm = (tt >> 2) * 16, tn = (tt & 3) * 16;
        v8f acc = {};
#pragma unroll
        for (int k0 = 0; k0 < DKc; k0 += 32) {
          v16bf a = frag_a_f32(&qmat[0][0], DKc + 1, tm, k0, nullptr);
          v16bf bb = frag_bT_f32(&kmat[0][0], DKc + 1, k0, tn);
          WMMA_BF16(acc, a, bb);
        }
#pragma unroll
        for (int i = 0; i < 8; ++i) {
          int m = tm + i + 8 * hihalf, n = tn + nloc;
          Tm[m][n] = acc[i] * Dm[m][n];
        }
      }
    }
    __syncthreads();

    // ---- out = (q*exp(g)) @ S + attn_c @ v_new -> global -------------------
    {
#pragma unroll
      for (int t = 0; t < 4; ++t) {
        int tt = wid * 4 + t;
        int tm = (tt >> 3) * 16, tn = (tt & 7) * 16;
        v8f acc = {};
#pragma unroll
        for (int k0 = 0; k0 < DKc; k0 += 32) {
          v16bf a = frag_a_f32(&qmat[0][0], DKc + 1, tm, k0, scl2);
          v16bf bb = frag_b_f32(&Sst[0][0], DVc + 1, k0, tn, nullptr);
          WMMA_BF16(acc, a, bb);
        }
#pragma unroll
        for (int k0 = 0; k0 < CH; k0 += 32) {
          v16bf a = frag_a_f32(&Tm[0][0], CH + 1, tm, k0, nullptr);
          v16bf bb = frag_b_f32(&vmat[0][0], DKc + 1, k0, tn, nullptr);
          WMMA_BF16(acc, a, bb);
        }
#pragma unroll
        for (int i = 0; i < 8; ++i) {
          int m = tm + i + 8 * hihalf, n = tn + nloc;
          core_out[((size_t)(b * Sseq + sbase + m) * HVc + h) * DVc + n] = acc[i];
        }
      }
    }

    // ---- S = S*exp(g_last) + kdec^T @ v_new --------------------------------
    if (tid < CH) scl1[tid] = __expf(gcum[CH - 1] - gcum[tid]);  // kdec k-scale
    __syncthreads();
    {
      v8f accS[8] = {};
#pragma unroll
      for (int t = 0; t < 8; ++t) {
        int tt = wid * 8 + t;
        int tm = (tt >> 3) * 16, tn = (tt & 7) * 16;
#pragma unroll
        for (int k0 = 0; k0 < CH; k0 += 32) {
          v16bf a = frag_aT_f32(&kmat[0][0], DKc + 1, tm, k0, scl1);
          v16bf bb = frag_b_f32(&vmat[0][0], DKc + 1, k0, tn, nullptr);
          WMMA_BF16(accS[t], a, bb);
        }
      }
      float eg = __expf(gcum[CH - 1]);
      __syncthreads();
#pragma unroll
      for (int t = 0; t < 8; ++t) {
        int tt = wid * 8 + t;
        int tm = (tt >> 3) * 16, tn = (tt & 7) * 16;
#pragma unroll
        for (int i = 0; i < 8; ++i) {
          int m = tm + i + 8 * hihalf, n = tn + nloc;
          Sst[m][n] = Sst[m][n] * eg + accS[t][i];
        }
      }
    }
    __syncthreads();
  }
}

// ---------------------------------------------------------------------------
// Kernel: RMS-norm (over DV) * norm_weight * silu(z) -> bf16.
// ---------------------------------------------------------------------------
__global__ __launch_bounds__(128) void norm_gate_kernel(const float* __restrict__ core,
                                                        const float* __restrict__ z,
                                                        const float* __restrict__ normw,
                                                        __bf16* __restrict__ out) {
  const int row = blockIdx.x;            // (b*S + s)*HV + h
  const float* cr = core + (size_t)row * DVc;
  const float* zr = z + (size_t)row * DVc;  // same flat offset (VAL = HV*DV)
  const int d = threadIdx.x;
  float v = cr[d];
  __shared__ float red[128];
  red[d] = v * v;
  __syncthreads();
  for (int st = 64; st > 0; st >>= 1) {
    if (d < st) red[d] += red[d + st];
    __syncthreads();
  }
  float var = red[0] * (1.f / 128.f);
  float scl = rsqrtf(var + EPSc) * normw[d];
  float zv = zr[d];
  float sz = zv / (1.f + __expf(-zv));
  out[(size_t)row * DVc + d] = (__bf16)(v * scl * sz);
}

// ---------------------------------------------------------------------------
// Host launcher
// ---------------------------------------------------------------------------
extern "C" void kernel_launch(void* const* d_in, const int* in_sizes, int n_in,
                              void* d_out, int out_size, void* d_ws, size_t ws_size,
                              hipStream_t stream) {
  const float* hs     = (const float*)d_in[0];
  const float* w_qkv  = (const float*)d_in[1];
  const float* w_z    = (const float*)d_in[2];
  const float* w_b    = (const float*)d_in[3];
  const float* w_a    = (const float*)d_in[4];
  const float* conv_w = (const float*)d_in[5];
  const float* dt_b   = (const float*)d_in[6];
  const float* A_log  = (const float*)d_in[7];
  const float* norm_w = (const float*)d_in[8];
  const float* w_out  = (const float*)d_in[9];
  float* out = (float*)d_out;
  (void)in_sizes; (void)n_in; (void)out_size; (void)ws_size;

  size_t off = 0;
  auto take = [&](size_t bytes) -> char* {
    char* p = (char*)d_ws + off;
    off += (bytes + 255) & ~(size_t)255;
    return p;
  };
  __bf16* hs_bf   = (__bf16*)take((size_t)MTOK * HIDc * 2);
  __bf16* wqkv_bf = (__bf16*)take((size_t)CDIM * HIDc * 2);
  __bf16* wz_bf   = (__bf16*)take((size_t)VALD * HIDc * 2);
  __bf16* wout_bf = (__bf16*)take((size_t)HIDc * VALD * 2);
  __bf16* qkv_bf  = (__bf16*)take((size_t)MTOK * CDIM * 2);
  __bf16* x_bf    = (__bf16*)take((size_t)MTOK * CDIM * 2);
  float*  z_f     = (float*)take((size_t)MTOK * VALD * 4);
  float*  g_f     = (float*)take((size_t)MTOK * HVc * 4);
  float*  beta_f  = (float*)take((size_t)MTOK * HVc * 4);
  float*  core_f  = (float*)take((size_t)MTOK * VALD * 4);
  __bf16* core_bf = (__bf16*)take((size_t)MTOK * VALD * 2);

  // 1) fp32 -> bf16 conversions
  {
    size_t n;
    n = (size_t)MTOK * HIDc;
    cvt_f32_bf16_kernel<<<(unsigned)(n / 4 / 256), 256, 0, stream>>>(hs, hs_bf, n);
    n = (size_t)CDIM * HIDc;
    cvt_f32_bf16_kernel<<<(unsigned)(n / 4 / 256), 256, 0, stream>>>(w_qkv, wqkv_bf, n);
    n = (size_t)VALD * HIDc;
    cvt_f32_bf16_kernel<<<(unsigned)(n / 4 / 256), 256, 0, stream>>>(w_z, wz_bf, n);
    n = (size_t)HIDc * VALD;
    cvt_f32_bf16_kernel<<<(unsigned)(n / 4 / 256), 256, 0, stream>>>(w_out, wout_bf, n);
  }

  // 2) qkv = hs @ w_qkv^T  (bf16 out), z = hs @ w_z^T (f32 out)
  {
    dim3 g1(CDIM / GBN, MTOK / GBM);
    gemm_bf16_kernel<<<g1, 256, 0, stream>>>(hs_bf, wqkv_bf, nullptr, qkv_bf,
                                             MTOK, CDIM, HIDc, 1);
    dim3 g2(VALD / GBN, MTOK / GBM);
    gemm_bf16_kernel<<<g2, 256, 0, stream>>>(hs_bf, wz_bf, z_f, nullptr,
                                             MTOK, VALD, HIDc, 0);
  }

  // 3) beta, g
  proj_ba_kernel<<<MTOK, 256, 0, stream>>>(hs, w_b, w_a, dt_b, A_log, beta_f, g_f);

  // 4) depthwise causal conv + SiLU
  {
    size_t nth = (size_t)Bc * CDIM * (Sseq / 8);
    conv_silu_kernel<<<(unsigned)(nth / 256), 256, 0, stream>>>(qkv_bf, conv_w, x_bf);
  }

  // 5) gated delta rule core
  core_kernel<<<Bc * HVc, 256, 0, stream>>>(x_bf, g_f, beta_f, core_f);

  // 6) norm + silu(z) gate -> bf16
  norm_gate_kernel<<<MTOK * HVc, 128, 0, stream>>>(core_f, z_f, norm_w, core_bf);

  // 7) out = core @ w_out^T (f32 out)
  {
    dim3 g3(HIDc / GBN, MTOK / GBM);
    gemm_bf16_kernel<<<g3, 256, 0, stream>>>(core_bf, wout_bf, out, nullptr,
                                             MTOK, HIDc, VALD, 0);
  }
}
